// GraphDiscriminator_14757507629468
// MI455X (gfx1250) — matrix operand order repaired and checked by reading, not compiled, not run on hardware
//
#include <hip/hip_runtime.h>

// GAT discriminator for MI455X (gfx1250, wave32).
#define N_NODES 20000
#define N_EDGES 320000
#define XD 128
#define RD 64
#define H_HEADS 8
#define L_LAYERS 3
#define ET (N_EDGES + N_NODES)   // edges + self loops = 340000
#define NEG_SLOPE 0.2f

typedef __attribute__((ext_vector_type(16))) __bf16 v16bf;
typedef __attribute__((ext_vector_type(8)))  float  v8f;

__device__ __forceinline__ unsigned int bfbits(float f) {
  unsigned int u = __builtin_bit_cast(unsigned int, f);
  return (u + 0x7FFFu + ((u >> 16) & 1u)) >> 16;   // round-to-nearest-even bf16
}

// ---------------------------------------------------------------------------
// Pre-pack A[M,K] (f32) into bf16 WMMA-A fragments.
// Layout: out_u32[((rt*nks + ks)*32 + lane)*8 + v] = {A[row][k0], A[row][k0+1]}
// with row = rt*16 + (lane&15), k0 = ks*32 + ((v&4)<<2) + ((lane>>4)<<3) + ((v&3)<<1)
// (CDNA5 ISA 16-bit A 16x32 layout, wave32).
// ---------------------------------------------------------------------------
__global__ void pack_a_bf16(const float* __restrict__ A,
                            unsigned int* __restrict__ out, int M, int K) {
  int idx = blockIdx.x * blockDim.x + threadIdx.x;
  int nks = K >> 5;
  int total = (M >> 4) * nks * 256;
  if (idx >= total) return;
  int v = idx & 7;
  int lane = (idx >> 3) & 31;
  int t = idx >> 8;                 // rt*nks + ks
  int rt = t / nks, ks = t % nks;
  int row = (rt << 4) + (lane & 15);
  int col = (ks << 5) + ((v & 4) << 2) + ((lane >> 4) << 3) + ((v & 3) << 1);
  const float* p = A + (size_t)row * K + col;
  out[idx] = bfbits(p[0]) | (bfbits(p[1]) << 16);
}

// Pre-pack B[K,Nc] (f32) into bf16 WMMA-B fragments.
// out_u32[((ct*nks + ks)*32 + lane)*8 + v] = {B[krow][col], B[krow+1][col]}
// with col = ct*16 + (lane&15), krow = ks*32 + ((v&4)<<2) + ((lane>>4)<<3) + ((v&3)<<1).
__global__ void pack_b_bf16(const float* __restrict__ B,
                            unsigned int* __restrict__ out, int K, int Nc) {
  int idx = blockIdx.x * blockDim.x + threadIdx.x;
  int nks = K >> 5;
  int total = (Nc >> 4) * nks * 256;
  if (idx >= total) return;
  int v = idx & 7;
  int lane = (idx >> 3) & 31;
  int t = idx >> 8;                 // ct*nks + ks
  int ct = t / nks, ks = t % nks;
  int col = (ct << 4) + (lane & 15);
  int krow = (ks << 5) + ((v & 4) << 2) + ((lane >> 4) << 3) + ((v & 3) << 1);
  out[idx] = bfbits(B[(size_t)krow * Nc + col]) |
             (bfbits(B[(size_t)(krow + 1) * Nc + col]) << 16);
}

// ---------------------------------------------------------------------------
// C[M,Nc] = A x B from pre-packed bf16 fragments. One wave per
// (16-row tile) x (COLS 16-col tiles); K fully unrolled. Inner loop is
// b128 loads + v_wmma_f32_16x16x32_bf16 only.
// ---------------------------------------------------------------------------
template <int K, int COLS>
__global__ __launch_bounds__(256) void gemm_wmma_packed(
    const __bf16* __restrict__ Ap, const __bf16* __restrict__ Bp,
    float* __restrict__ C, int M, int Nc) {
  constexpr int NKS = K / 32;
  int wid  = (blockIdx.x * blockDim.x + threadIdx.x) >> 5;
  int lane = threadIdx.x & 31;
  int colGroups = (Nc >> 4) / COLS;
  int waves = (M >> 4) * colGroups;
  if (wid >= waves) return;               // wave-uniform exit
  int rt = wid / colGroups;
  int cg = wid % colGroups;
  v8f acc[COLS] = {};
#pragma unroll
  for (int ks = 0; ks < NKS; ++ks) {
    v16bf av = *(const v16bf*)(Ap + (((size_t)rt * NKS + ks) * 32 + lane) * 16);
#pragma unroll
    for (int c = 0; c < COLS; ++c) {
      int ct = cg * COLS + c;
      v16bf bv = *(const v16bf*)(Bp + (((size_t)ct * NKS + ks) * 32 + lane) * 16);
      acc[c] = __builtin_amdgcn_wmma_f32_16x16x32_bf16(
          false, av, false, bv, (short)0, acc[c], false, false);
    }
  }
  int mo = (lane >> 4) << 3, ncol = lane & 15;
#pragma unroll
  for (int c = 0; c < COLS; ++c) {
    int colBase = (cg * COLS + c) << 4;
#pragma unroll
    for (int v = 0; v < 8; ++v)
      C[(size_t)((rt << 4) + mo + v) * Nc + colBase + ncol] = acc[c][v];
  }
}

// alpha_s[n,h] = <hproj[n,h,:], att_s[h,:]> ; same for alpha_d.
// One wave32 per (node, head); butterfly shuffle reduction.
__global__ __launch_bounds__(256) void gat_alphas(
    const float* __restrict__ hproj, const float* __restrict__ att_s,
    const float* __restrict__ att_d, float* __restrict__ alpha_s,
    float* __restrict__ alpha_d) {
  int n = blockIdx.x;
  int w = threadIdx.x >> 5;        // head 0..7
  int lane = threadIdx.x & 31;
  const float* hp = hproj + (size_t)n * (H_HEADS * RD) + w * RD;
  float v0 = hp[lane], v1 = hp[lane + 32];
  float s = v0 * att_s[w * RD + lane] + v1 * att_s[w * RD + lane + 32];
  float d = v0 * att_d[w * RD + lane] + v1 * att_d[w * RD + lane + 32];
#pragma unroll
  for (int off = 16; off > 0; off >>= 1) {
    s += __shfl_xor(s, off, 32);
    d += __shfl_xor(d, off, 32);
  }
  if (lane == 0) {
    alpha_s[n * H_HEADS + w] = s;
    alpha_d[n * H_HEADS + w] = d;
  }
}

__global__ void gat_init(float* __restrict__ smax, float* __restrict__ denom,
                         float* __restrict__ accum) {
  int i = blockIdx.x * blockDim.x + threadIdx.x;
  if (i < N_NODES * RD) accum[i] = 0.f;
  if (i < N_NODES * H_HEADS) { smax[i] = -3.0e38f; denom[i] = 0.f; }
}

__device__ __forceinline__ void atomicMaxF(float* addr, float val) {
  if (val >= 0.f) atomicMax((int*)addr, __float_as_int(val));
  else            atomicMin((unsigned int*)addr, __float_as_uint(val));
}

__global__ void gat_score_max(const float* __restrict__ alpha_s,
                              const float* __restrict__ alpha_d,
                              const int* __restrict__ e,
                              float* __restrict__ score,
                              float* __restrict__ smax) {
  int gid = blockIdx.x * blockDim.x + threadIdx.x;
  if (gid >= ET * H_HEADS) return;
  int j = gid >> 3, h = gid & 7;
  int s, d;
  if (j < N_EDGES) { s = e[j]; d = e[N_EDGES + j]; }
  else             { s = d = j - N_EDGES; }          // self loop
  float sc = alpha_s[s * H_HEADS + h] + alpha_d[d * H_HEADS + h];
  sc = sc > 0.f ? sc : NEG_SLOPE * sc;               // leaky relu
  score[gid] = sc;
  atomicMaxF(&smax[d * H_HEADS + h], sc);
}

__global__ void gat_exp_sum(float* __restrict__ score,
                            const float* __restrict__ smax,
                            float* __restrict__ denom,
                            const int* __restrict__ e) {
  int gid = blockIdx.x * blockDim.x + threadIdx.x;
  if (gid >= ET * H_HEADS) return;
  int j = gid >> 3, h = gid & 7;
  int d = (j < N_EDGES) ? e[N_EDGES + j] : j - N_EDGES;
  float v = __expf(score[gid] - smax[d * H_HEADS + h]);
  score[gid] = v;                                    // buffer reused as ex
  atomicAdd(&denom[d * H_HEADS + h], v);
}

// Per edge: accum[dst,c] += (1/H) * sum_h alpha[e,h] * hproj[src,h,c]
// 64 threads per edge (channels), 4 edges per 256-thread block.
// Head-mean folded in => 8x less atomic traffic than reference order.
__global__ __launch_bounds__(256) void gat_aggregate(
    const float* __restrict__ hproj, const float* __restrict__ ex,
    const float* __restrict__ denom, const int* __restrict__ e,
    float* __restrict__ accum) {
  __shared__ float sal[4][H_HEADS];
  int le = threadIdx.x >> 6;   // local edge 0..3
  int c  = threadIdx.x & 63;   // channel
  int j = blockIdx.x * 4 + le;
  int s = 0, d = 0;
  bool ok = j < ET;
  if (ok) {
    if (j < N_EDGES) { s = e[j]; d = e[N_EDGES + j]; }
    else             { s = d = j - N_EDGES; }
    if (c < H_HEADS)
      sal[le][c] = ex[j * H_HEADS + c] /
                   (denom[d * H_HEADS + c] + 1e-16f) * (1.f / H_HEADS);
  }
  __syncthreads();
  if (!ok) return;
  const float* hp = hproj + (size_t)s * (H_HEADS * RD) + c;
  float a = 0.f;
#pragma unroll
  for (int h = 0; h < H_HEADS; ++h) a += sal[le][h] * hp[h * RD];
  atomicAdd(&accum[d * RD + c], a);
}

__global__ void gat_finalize(const float* __restrict__ accum,
                             const float* __restrict__ bias,
                             float* __restrict__ hout) {
  int i = blockIdx.x * blockDim.x + threadIdx.x;
  if (i >= N_NODES * RD) return;
  float v = accum[i] + bias[i & 63];
  hout[i] = v > 0.f ? v : 0.f;                       // relu
}

__global__ void zero_pooled(float* __restrict__ pooled) {
  if (threadIdx.x < 2 * RD) pooled[threadIdx.x] = 0.f;
}

__global__ __launch_bounds__(256) void pool_sum(const float* __restrict__ src,
                                                float* __restrict__ dst) {
  int c = threadIdx.x & 63;
  int part = threadIdx.x >> 6;
  float s = 0.f;
  for (int n = blockIdx.x * 4 + part; n < N_NODES; n += gridDim.x * 4)
    s += src[(size_t)n * RD + c];
  atomicAdd(&dst[c], s);
}

__global__ void final_fc(const float* __restrict__ pooled,
                         const float* __restrict__ fc_w,
                         const float* __restrict__ fc_b,
                         float* __restrict__ out) {
  if (threadIdx.x == 0 && blockIdx.x == 0) {
    float z = fc_b[0];
    for (int i = 0; i < 2 * RD; ++i) z += pooled[i] * fc_w[i];
    out[0] = 1.f / (1.f + __expf(-z));
  }
}

extern "C" void kernel_launch(void* const* d_in, const int* in_sizes, int n_in,
                              void* d_out, int out_size, void* d_ws, size_t ws_size,
                              hipStream_t stream) {
  const float* x       = (const float*)d_in[0];
  const int*   e       = (const int*)d_in[1];      // [2,E] int32
  const float* cfeat   = (const float*)d_in[2];
  const float* W_embed = (const float*)d_in[3];
  const float* Ws      = (const float*)d_in[4];    // [L,RD,H*RD]
  const float* att_src = (const float*)d_in[5];    // [L,H,RD]
  const float* att_dst = (const float*)d_in[6];
  const float* conv_b  = (const float*)d_in[7];    // [L,RD]
  const float* fc_w    = (const float*)d_in[8];    // [2*RD]
  const float* fc_b    = (const float*)d_in[9];
  float* out = (float*)d_out;

  // Workspace layout (~71 MB), 32B-aligned chunks (offsets in f32 units).
  float* ws = (float*)d_ws;
  size_t off = 0;
  float* h_cur   = ws + off; off += (size_t)N_NODES * RD;            // 1.28M
  float* hproj   = ws + off; off += (size_t)N_NODES * H_HEADS * RD;  // 10.24M
  float* alpha_s = ws + off; off += N_NODES * H_HEADS;
  float* alpha_d = ws + off; off += N_NODES * H_HEADS;
  float* smax    = ws + off; off += N_NODES * H_HEADS;
  float* denom   = ws + off; off += N_NODES * H_HEADS;
  float* exbuf   = ws + off; off += (size_t)ET * H_HEADS;            // 2.72M
  float* accum   = ws + off; off += (size_t)N_NODES * RD;
  float* pooled  = ws + off; off += 2 * RD;
  unsigned int* apack = (unsigned int*)(ws + off);                   // bf16 A frags
  off += (size_t)N_NODES * XD / 2;        // max M*K/2 u32 (embed: 1.28M u32)
  unsigned int* bpack = (unsigned int*)(ws + off);
  off += (RD * H_HEADS * RD) / 2;         // max K*Nc/2 u32 (16K u32)

  // ---- embed: h = x @ W_embed  (20000x128 @ 128x64) ----
  {
    int atot = (N_NODES / 16) * (XD / 32) * 256;           // 1.28M
    pack_a_bf16<<<(atot + 255) / 256, 256, 0, stream>>>(x, apack, N_NODES, XD);
    int btot = (RD / 16) * (XD / 32) * 256;                // 4096
    pack_b_bf16<<<(btot + 255) / 256, 256, 0, stream>>>(W_embed, bpack, XD, RD);
    int waves = (N_NODES / 16) * ((RD / 16) / 4);          // 1250
    gemm_wmma_packed<XD, 4><<<(waves * 32 + 255) / 256, 256, 0, stream>>>(
        (const __bf16*)apack, (const __bf16*)bpack, h_cur, N_NODES, RD);
  }

  int eth = ET * H_HEADS;
  for (int l = 0; l < L_LAYERS; ++l) {
    // ---- hproj = h_cur @ Ws[l]  (20000x64 @ 64x512) ----
    int atot = (N_NODES / 16) * (RD / 32) * 256;           // 640K
    pack_a_bf16<<<(atot + 255) / 256, 256, 0, stream>>>(h_cur, apack, N_NODES, RD);
    int btot = ((H_HEADS * RD) / 16) * (RD / 32) * 256;    // 16384
    pack_b_bf16<<<(btot + 255) / 256, 256, 0, stream>>>(
        Ws + (size_t)l * RD * H_HEADS * RD, bpack, RD, H_HEADS * RD);
    int waves = (N_NODES / 16) * (((H_HEADS * RD) / 16) / 4);  // 10000
    gemm_wmma_packed<RD, 4><<<(waves * 32 + 255) / 256, 256, 0, stream>>>(
        (const __bf16*)apack, (const __bf16*)bpack, hproj, N_NODES, H_HEADS * RD);

    gat_alphas<<<N_NODES, 256, 0, stream>>>(
        hproj, att_src + l * H_HEADS * RD, att_dst + l * H_HEADS * RD,
        alpha_s, alpha_d);
    gat_init<<<(N_NODES * RD + 255) / 256, 256, 0, stream>>>(smax, denom, accum);
    gat_score_max<<<(eth + 255) / 256, 256, 0, stream>>>(
        alpha_s, alpha_d, e, exbuf, smax);
    gat_exp_sum<<<(eth + 255) / 256, 256, 0, stream>>>(exbuf, smax, denom, e);
    gat_aggregate<<<(ET + 3) / 4, 256, 0, stream>>>(hproj, exbuf, denom, e, accum);
    gat_finalize<<<(N_NODES * RD + 255) / 256, 256, 0, stream>>>(
        accum, conv_b + l * RD, h_cur);
  }

  zero_pooled<<<1, 128, 0, stream>>>(pooled);
  pool_sum<<<512, 256, 0, stream>>>(h_cur, pooled);
  pool_sum<<<512, 256, 0, stream>>>(cfeat, pooled + RD);
  final_fc<<<1, 32, 0, stream>>>(pooled, fc_w, fc_b, out);

  (void)in_sizes; (void)n_in; (void)out_size; (void)ws_size;
}
